// PreNormTransformer_7997229105672
// MI455X (gfx1250) — matrix operand order repaired and checked by reading, not compiled, not run on hardware
//
#include <hip/hip_runtime.h>

// ---------------------------------------------------------------------------
// PreNorm transformer block for MI455X (gfx1250, wave32, WMMA).
// All GEMMs (QKV/WO/W1/W3/W2) and both attention matmuls use
// v_wmma_f32_16x16x32_bf16 (bf16 in, f32 accumulate).
// GEMM: 128x128 tile / 256 threads / BK=64 -> 16 WMMA per wave per stage.
// Tile staging uses GLOBAL_LOAD_ASYNC_TO_LDS (ASYNCcnt) when available.
// ---------------------------------------------------------------------------

#define BATCH   2
#define SEQ     2048
#define DMODEL  1024
#define NHEADS  16
#define DK      64
#define DFF     2816
#define MTOK    (BATCH * SEQ)   // 4096 token rows

typedef float  v8f   __attribute__((ext_vector_type(8)));
typedef __bf16 v16bf __attribute__((ext_vector_type(16)));
typedef int    v4i   __attribute__((ext_vector_type(4)));

#if defined(__has_builtin)
#if __has_builtin(__builtin_amdgcn_global_load_async_to_lds_b128)
#define HAVE_ASYNC_LDS 1
#endif
#endif

#ifdef HAVE_ASYNC_LDS
typedef __attribute__((address_space(3))) v4i as3_v4i;
#endif

struct Frag {
    union { uint4 u[2]; v16bf v; };
};

static __device__ inline v8f wmma_bf16(v16bf a, v16bf b, v8f c) {
    // D = A(16x32) * B(32x16) + C(16x16 f32)
    return __builtin_amdgcn_wmma_f32_16x16x32_bf16(
        /*neg_a=*/false, a, /*neg_b=*/false, b,
        /*c_mod=*/(short)0, c, /*reuse_a=*/false, /*reuse_b=*/false);
}

// A-matrix fragment (16x32 bf16), ISA 7.12.2 layout:
// lane m (0..15): K = 0..7 (v0..3), 16..23 (v4..7); lane m+16: K = 8..15, 24..31
static __device__ inline void load_a_frag(Frag& f, const __bf16* base, int ld,
                                          int row0, int col0, int lane) {
    int lr = lane & 15, hi = lane >> 4;
    const __bf16* rp = base + (size_t)(row0 + lr) * ld + col0;
    f.u[0] = *(const uint4*)(rp + hi * 8);
    f.u[1] = *(const uint4*)(rp + 16 + hi * 8);
}

// B-matrix fragment (32x16 bf16) from a row-major [N][K] tile (i.e. B^T):
// lane holds column N = lane%16; lanes 0-15 K=0..7,16..23, lanes 16-31 K=8..15,24..31
static __device__ inline void load_b_frag(Frag& f, const __bf16* base, int ld,
                                          int row0, int col0, int lane) {
    int lr = lane & 15, hi = lane >> 4;
    const __bf16* rp = base + (size_t)(row0 + lr) * ld + col0 + hi * 8;
    f.u[0] = *(const uint4*)(rp);
    f.u[1] = *(const uint4*)(rp + 16);
}

// ---------------------------------------------------------------------------
// fp32 -> bf16 convert (weights)
// ---------------------------------------------------------------------------
__global__ __launch_bounds__(256) void f32_to_bf16_kernel(const float* __restrict__ in,
                                                          __bf16* __restrict__ out, int n) {
    int i = blockIdx.x * 256 + threadIdx.x;
    if (i < n) out[i] = (__bf16)in[i];
}

// ---------------------------------------------------------------------------
// RMSNorm: one block (256 threads) per token row of 1024
// ---------------------------------------------------------------------------
__global__ __launch_bounds__(256) void rmsnorm_kernel(const float* __restrict__ x,
                                                      const float* __restrict__ g,
                                                      __bf16* __restrict__ out) {
    int tkn = blockIdx.x, t = threadIdx.x;
    const float* xr = x + (size_t)tkn * DMODEL;
    float vals[4], ss = 0.f;
#pragma unroll
    for (int i = 0; i < 4; i++) { vals[i] = xr[t + i * 256]; ss += vals[i] * vals[i]; }
#pragma unroll
    for (int off = 16; off; off >>= 1) ss += __shfl_xor(ss, off);
    __shared__ float red[8];
    if ((t & 31) == 0) red[t >> 5] = ss;
    __syncthreads();
    float tot = 0.f;
#pragma unroll
    for (int i = 0; i < 8; i++) tot += red[i];
    float inv = rsqrtf(tot * (1.0f / DMODEL) + 1e-5f);
#pragma unroll
    for (int i = 0; i < 4; i++) {
        int c = t + i * 256;
        out[(size_t)tkn * DMODEL + c] = (__bf16)(vals[i] * g[c] * inv);
    }
}

// ---------------------------------------------------------------------------
// RoPE (interleaved pairs), in place on bf16 projection buffer
// ---------------------------------------------------------------------------
__global__ __launch_bounds__(256) void rope_kernel(__bf16* __restrict__ buf) {
    int idx = blockIdx.x * 256 + threadIdx.x;          // over MTOK*512 pairs
    if (idx >= MTOK * (DMODEL / 2)) return;
    int row  = idx >> 9;                               // token row
    int p    = idx & 511;
    int head = p >> 5, i = p & 31;
    int s    = row & (SEQ - 1);
    // theta^(-2i/dk) = exp(-i * ln(10000)/32)
    float ang = (float)s * __expf(-0.28782313662425572f * (float)i);
    float sn, cs;
    __sincosf(ang, &sn, &cs);
    size_t base = (size_t)row * DMODEL + head * DK + 2 * i;
    float e = (float)buf[base], o = (float)buf[base + 1];
    buf[base]     = (__bf16)(e * cs - o * sn);
    buf[base + 1] = (__bf16)(o * cs + e * sn);
}

// ---------------------------------------------------------------------------
// Generic NT GEMM: C[M,N] = A[M,K] (bf16, row-major) * B[N,K]^T (bf16, row-major)
// 128x128 tile per 8-wave workgroup, k-step 64 through LDS, WMMA bf16.
// Each wave: 64x32 patch = 4x2 accumulators, 16 WMMA per k-stage.
// MODE 0: store bf16           MODE 1: store f32 + residual
// MODE 2: store bf16 = silu(u) * acc   (SwiGLU fusion, u read at same index)
// ---------------------------------------------------------------------------
template <int MODE>
__global__ __launch_bounds__(256) void gemm_nt_kernel(const __bf16* __restrict__ A,
                                                      const __bf16* __restrict__ Bw,
                                                      int M, int N, int K,
                                                      float* __restrict__ outf,
                                                      const float* __restrict__ resid,
                                                      __bf16* __restrict__ outb,
                                                      const __bf16* __restrict__ ugate) {
    __shared__ __bf16 a_tile[128][64];
    __shared__ __bf16 b_tile[128][64];

    int mbase = blockIdx.y * 128, nbase = blockIdx.x * 128;
    int t = threadIdx.x, lane = t & 31, w = t >> 5;
    int wm = (w >> 2) * 64;        // wave row base   (2 wave-rows of 64)
    int wn = (w & 3) * 32;         // wave col base   (4 wave-cols of 32)

    v8f acc[4][2] = {};

    int srow = t >> 1, scol = (t & 1) * 32;   // staging: 64B per thread per tile
    const __bf16* ag = A  + (size_t)(mbase + srow) * K + scol;
    const __bf16* bg = Bw + (size_t)(nbase + srow) * K + scol;

    for (int k0 = 0; k0 < K; k0 += 64) {
        __syncthreads();
#ifdef HAVE_ASYNC_LDS
        {   // CDNA5 async copy: global -> LDS without VGPR round trip (ASYNCcnt)
            v4i* ga = (v4i*)ag;
            v4i* gb = (v4i*)bg;
            as3_v4i* la = (as3_v4i*)&a_tile[srow][scol];
            as3_v4i* lb = (as3_v4i*)&b_tile[srow][scol];
#pragma unroll
            for (int q = 0; q < 4; q++) {
                __builtin_amdgcn_global_load_async_to_lds_b128(ga + q, la + q, 0, 0);
                __builtin_amdgcn_global_load_async_to_lds_b128(gb + q, lb + q, 0, 0);
            }
        }
#if __has_builtin(__builtin_amdgcn_s_wait_asynccnt)
        __builtin_amdgcn_s_wait_asynccnt(0);
#else
        asm volatile("s_wait_asynccnt 0x0" ::: "memory");
#endif
#else
        // synchronous staging fallback (64B per thread per tile)
#pragma unroll
        for (int q = 0; q < 4; q++) {
            *(uint4*)&a_tile[srow][scol + q * 8] = *(const uint4*)(ag + q * 8);
            *(uint4*)&b_tile[srow][scol + q * 8] = *(const uint4*)(bg + q * 8);
        }
#endif
        if (k0 + 64 < K) {                       // hint next k-tile into cache
            __builtin_prefetch(ag + 64, 0, 0);
            __builtin_prefetch(bg + 64, 0, 0);
        }
        ag += 64; bg += 64;
        __syncthreads();

#pragma unroll
        for (int kc = 0; kc < 64; kc += 32) {
            Frag af[4], bf[2];
#pragma unroll
            for (int i = 0; i < 4; i++)
                load_a_frag(af[i], &a_tile[0][0], 64, wm + i * 16, kc, lane);
#pragma unroll
            for (int j = 0; j < 2; j++)
                load_b_frag(bf[j], &b_tile[0][0], 64, wn + j * 16, kc, lane);
#pragma unroll
            for (int i = 0; i < 4; i++)
#pragma unroll
                for (int j = 0; j < 2; j++)
                    acc[i][j] = wmma_bf16(af[i].v, bf[j].v, acc[i][j]);
        }
    }

    int hi = lane >> 4, nl = lane & 15;
#pragma unroll
    for (int i = 0; i < 4; i++)
#pragma unroll
        for (int j = 0; j < 2; j++)
#pragma unroll
            for (int v = 0; v < 8; v++) {
                int mg = mbase + wm + i * 16 + v + 8 * hi;
                int ng = nbase + wn + j * 16 + nl;
                size_t idx = (size_t)mg * N + ng;
                float val = acc[i][j][v];
                if (MODE == 0) {
                    outb[idx] = (__bf16)val;
                } else if (MODE == 1) {
                    outf[idx] = val + resid[idx];
                } else {
                    float u = (float)ugate[idx];
                    float s = u / (1.f + __expf(-u));   // silu(u)
                    outb[idx] = (__bf16)(s * val);
                }
            }
}

// ---------------------------------------------------------------------------
// Flash attention (causal). Grid: (S/64, B*H). Block: 128 (4 waves).
// Each wave owns 16 query rows; K/V tiles (32 keys) staged in LDS per block.
// Q*K^T and P*V both via WMMA bf16; online softmax on the f32 accumulators.
// ---------------------------------------------------------------------------
__global__ __launch_bounds__(128) void flash_attn_kernel(const __bf16* __restrict__ qb,
                                                         const __bf16* __restrict__ kb,
                                                         const __bf16* __restrict__ vb,
                                                         __bf16* __restrict__ ob) {
    __shared__ __bf16 k_tile[32][64];        // [key][dk]
    __shared__ __bf16 vt_tile[64][32];       // [dk][key]   (V transposed)
    __shared__ __bf16 p_tile[4][16][32];     // per-wave P staging

    int bh = blockIdx.y, b = bh >> 4, h = bh & 15;
    int t = threadIdx.x, lane = t & 31, w = t >> 5;
    int q0 = blockIdx.x * 64 + w * 16;       // first query row of this wave
    size_t tokbase = (size_t)b * SEQ;
    int lr = lane & 15, hi = lane >> 4;

    // Q fragments for dk halves, loaded straight from global (A layout)
    Frag qf[2];
    const __bf16* qbase = qb + (tokbase + q0) * DMODEL + h * DK;
#pragma unroll
    for (int f = 0; f < 2; f++) load_a_frag(qf[f], qbase, DMODEL, 0, f * 32, lane);

    v8f O[4] = {};
    float rmax[8], rsum[8];
#pragma unroll
    for (int v = 0; v < 8; v++) { rmax[v] = -1e30f; rsum[v] = 0.f; }

    int kend = blockIdx.x * 64 + 64;         // causal bound, uniform per block
    int skey = t >> 2, scol = (t & 3) * 16;  // cooperative staging coords

    for (int kb0 = 0; kb0 < kend; kb0 += 32) {
        __syncthreads();
        {   // stage K tile row-major and V tile transposed
            const __bf16* kg = kb + (tokbase + kb0 + skey) * DMODEL + h * DK + scol;
            *(uint4*)&k_tile[skey][scol]     = *(const uint4*)(kg);
            *(uint4*)&k_tile[skey][scol + 8] = *(const uint4*)(kg + 8);
            const __bf16* vg = vb + (tokbase + kb0 + skey) * DMODEL + h * DK + scol;
            union { uint4 u[2]; __bf16 e[16]; } vv;
            vv.u[0] = *(const uint4*)(vg);
            vv.u[1] = *(const uint4*)(vg + 8);
#pragma unroll
            for (int e = 0; e < 16; e++) vt_tile[scol + e][skey] = vv.e[e];
        }
        __syncthreads();

        // scores for 32 keys: two 16x16 tiles, K-loop over dk (2 x WMMA each)
        v8f sc[2] = {};
#pragma unroll
        for (int c = 0; c < 2; c++)
#pragma unroll
            for (int f = 0; f < 2; f++) {
                Frag kf;
                load_b_frag(kf, &k_tile[0][0], 64, c * 16, f * 32, lane);
                sc[c] = wmma_bf16(qf[f].v, kf.v, sc[c]);
            }

        // scale + causal mask
#pragma unroll
        for (int c = 0; c < 2; c++)
#pragma unroll
            for (int v = 0; v < 8; v++) {
                float s = sc[c][v] * 0.125f;             // dk^-0.5
                int qg = q0 + v + 8 * hi;
                int kg = kb0 + c * 16 + lr;
                sc[c][v] = (kg > qg) ? -1e30f : s;
            }

        // online softmax update (rows live across 16 lanes of a half-wave)
#pragma unroll
        for (int v = 0; v < 8; v++) {
            float tm = fmaxf(sc[0][v], sc[1][v]);
#pragma unroll
            for (int off = 8; off; off >>= 1) tm = fmaxf(tm, __shfl_xor(tm, off));
            float nm = fmaxf(rmax[v], tm);
            float scale = __expf(rmax[v] - nm);
            rmax[v] = nm;
            float p0 = __expf(sc[0][v] - nm);
            float p1 = __expf(sc[1][v] - nm);
            sc[0][v] = p0; sc[1][v] = p1;
            float ps = p0 + p1;
#pragma unroll
            for (int off = 8; off; off >>= 1) ps += __shfl_xor(ps, off);
            rsum[v] = rsum[v] * scale + ps;
#pragma unroll
            for (int j = 0; j < 4; j++) O[j][v] *= scale;
        }

        // P -> bf16 into per-wave LDS, then reload in A-fragment layout
#pragma unroll
        for (int v = 0; v < 8; v++) {
            p_tile[w][v + 8 * hi][lr]      = (__bf16)sc[0][v];
            p_tile[w][v + 8 * hi][16 + lr] = (__bf16)sc[1][v];
        }
        asm volatile("s_wait_dscnt 0" ::: "memory");
        Frag pf;
        load_a_frag(pf, &p_tile[w][0][0], 32, 0, 0, lane);

        // O += P (16x32) * V (32 x 64) over 4 column blocks
#pragma unroll
        for (int j = 0; j < 4; j++) {
            Frag vf;
            load_b_frag(vf, &vt_tile[0][0], 32, j * 16, 0, lane);
            O[j] = wmma_bf16(pf.v, vf.v, O[j]);
        }
    }

    // normalize and store attention output (bf16, [token][h*64+col])
#pragma unroll
    for (int j = 0; j < 4; j++)
#pragma unroll
        for (int v = 0; v < 8; v++) {
            size_t row = tokbase + q0 + v + 8 * hi;
            int col = h * DK + j * 16 + lr;
            ob[row * DMODEL + col] = (__bf16)(O[j][v] * (1.f / rsum[v]));
        }
}

// ---------------------------------------------------------------------------
// Host-side orchestration
// ---------------------------------------------------------------------------
extern "C" void kernel_launch(void* const* d_in, const int* in_sizes, int n_in,
                              void* d_out, int out_size, void* d_ws, size_t ws_size,
                              hipStream_t stream) {
    (void)in_sizes; (void)n_in; (void)out_size; (void)ws_size;

    const float* x  = (const float*)d_in[0];
    const float* g1 = (const float*)d_in[1];
    const float* g2 = (const float*)d_in[2];
    const float* WQ = (const float*)d_in[3];
    const float* WK = (const float*)d_in[4];
    const float* WV = (const float*)d_in[5];
    const float* WO = (const float*)d_in[6];
    const float* W1 = (const float*)d_in[7];
    const float* W2 = (const float*)d_in[8];
    const float* W3 = (const float*)d_in[9];
    float* out = (float*)d_out;

    // workspace carve (256B aligned)
    char* p = (char*)d_ws;
    size_t off = 0;
    auto take = [&](size_t bytes) -> char* {
        char* r = p + off;
        off += (bytes + 255) & ~(size_t)255;
        return r;
    };
    const size_t NDD  = (size_t)DMODEL * DMODEL;       // 1M
    const size_t NDF  = (size_t)DFF * DMODEL;          // 2.88M
    const size_t NTD  = (size_t)MTOK * DMODEL;         // 4.19M
    const size_t NTF  = (size_t)MTOK * DFF;            // 11.5M

    __bf16* wqb  = (__bf16*)take(NDD * 2);
    __bf16* wkb  = (__bf16*)take(NDD * 2);
    __bf16* wvb  = (__bf16*)take(NDD * 2);
    __bf16* wob  = (__bf16*)take(NDD * 2);
    __bf16* w1b  = (__bf16*)take(NDF * 2);
    __bf16* w2b  = (__bf16*)take(NDF * 2);
    __bf16* w3b  = (__bf16*)take(NDF * 2);
    __bf16* h1b  = (__bf16*)take(NTD * 2);
    __bf16* qbuf = (__bf16*)take(NTD * 2);
    __bf16* kbuf = (__bf16*)take(NTD * 2);
    __bf16* vbuf = (__bf16*)take(NTD * 2);
    __bf16* attb = (__bf16*)take(NTD * 2);
    float*  yf   = (float*) take(NTD * 4);
    __bf16* h2b  = (__bf16*)take(NTD * 2);
    __bf16* ub   = (__bf16*)take(NTF * 2);
    __bf16* ffb  = (__bf16*)take(NTF * 2);

    auto cvt = [&](const float* src, __bf16* dst, size_t n) {
        f32_to_bf16_kernel<<<dim3((unsigned)((n + 255) / 256)), dim3(256), 0, stream>>>(src, dst, (int)n);
    };
    cvt(WQ, wqb, NDD); cvt(WK, wkb, NDD); cvt(WV, wvb, NDD); cvt(WO, wob, NDD);
    cvt(W1, w1b, NDF); cvt(W2, w2b, NDF); cvt(W3, w3b, NDF);

    // h1 = rmsnorm(x, g1)
    rmsnorm_kernel<<<dim3(MTOK), dim3(256), 0, stream>>>(x, g1, h1b);

    dim3 gemmBlk(256);
    dim3 gridDD(DMODEL / 128, MTOK / 128);   // N=1024 -> (8, 32)
    dim3 gridDF(DFF / 128,    MTOK / 128);   // N=2816 -> (22, 32)

    // Q/K/V projections (bf16 out)
    gemm_nt_kernel<0><<<gridDD, gemmBlk, 0, stream>>>(h1b, wqb, MTOK, DMODEL, DMODEL, nullptr, nullptr, qbuf, nullptr);
    gemm_nt_kernel<0><<<gridDD, gemmBlk, 0, stream>>>(h1b, wkb, MTOK, DMODEL, DMODEL, nullptr, nullptr, kbuf, nullptr);
    gemm_nt_kernel<0><<<gridDD, gemmBlk, 0, stream>>>(h1b, wvb, MTOK, DMODEL, DMODEL, nullptr, nullptr, vbuf, nullptr);

    // RoPE on q, k
    int npair = MTOK * (DMODEL / 2);
    rope_kernel<<<dim3((npair + 255) / 256), dim3(256), 0, stream>>>(qbuf);
    rope_kernel<<<dim3((npair + 255) / 256), dim3(256), 0, stream>>>(kbuf);

    // causal flash attention
    flash_attn_kernel<<<dim3(SEQ / 64, BATCH * NHEADS), dim3(128), 0, stream>>>(qbuf, kbuf, vbuf, attb);

    // y = attn @ WO^T + x   (f32)
    gemm_nt_kernel<1><<<gridDD, gemmBlk, 0, stream>>>(attb, wob, MTOK, DMODEL, DMODEL, yf, x, nullptr, nullptr);

    // h2 = rmsnorm(y, g2)
    rmsnorm_kernel<<<dim3(MTOK), dim3(256), 0, stream>>>(yf, g2, h2b);

    // u = h2 @ W1^T (bf16);  ff_in = silu(u) * (h2 @ W3^T)
    gemm_nt_kernel<0><<<gridDF, gemmBlk, 0, stream>>>(h2b, w1b, MTOK, DFF, DMODEL, nullptr, nullptr, ub, nullptr);
    gemm_nt_kernel<2><<<gridDF, gemmBlk, 0, stream>>>(h2b, w3b, MTOK, DFF, DMODEL, nullptr, nullptr, ffb, ub);

    // out = ff_in @ W2^T + y
    gemm_nt_kernel<1><<<gridDD, gemmBlk, 0, stream>>>(ffb, w2b, MTOK, DMODEL, DFF, out, yf, nullptr, nullptr);
}
